// LinearAddmv_21251498180726
// MI455X (gfx1250) — compile-verified, matrix operand order checked by reading, not run on hardware
//
#include <hip/hip_runtime.h>

// ---------------------------------------------------------------------------
// out[B, OUT_F] = X[B, IN_F] . W[OUT_F, IN_F]^T + b[OUT_F]   (all fp32)
//
// bf16x3 split-precision GEMM (x = x_hi + x_lo, 3 WMMA terms: hi*hi + hi*lo
// + lo*hi) on v_wmma_f32_16x16x32_bf16. Near-fp32 accuracy (~1e-5 rel),
// ~2.7x faster than the native V_WMMA_F32_16X16X4_F32 path (K=32 vs K=4).
//
// Phase 1: memory-bound split pass writes X_hi/X_lo/W_hi/W_lo (bf16,
//          row-major) into d_ws (128 MB total; fits the 192 MB L2).
// Phase 2: compute-bound GEMM loads preconverted bf16 -> zero conversion
//          VALU in the k-loop; 24 b128 loads : 24 WMMAs per k-step per wave.
//          12 independent stream pointers precomputed outside the loop,
//          k-loop unrolled x4 so address bumps amortize via 24-bit offsets.
//
// Wave tile 32x64, block = 8 waves (2M x 4N) = 64x256 tile, grid 16x64.
// All dims divide exactly -> no predication, EXEC all-ones as WMMA requires.
// ---------------------------------------------------------------------------

typedef __attribute__((ext_vector_type(16))) __bf16 v16bf;
typedef __attribute__((ext_vector_type(8)))  __bf16 v8bf;
typedef __attribute__((ext_vector_type(8)))  float  v8f;

#define IN_F  4096
#define OUT_F 4096
#define BATCH 4096
#define NELEM ((size_t)4096 * 4096)

// ---------------------------------------------------------------------------
// Phase 1: fp32 -> (bf16 hi, bf16 lo) split. 8 elements / thread, b128 I/O.
// grid.y selects the matrix (0 = X, 1 = W).
// ---------------------------------------------------------------------------
__global__ __launch_bounds__(256)
void split_pass(const float* __restrict__ X, const float* __restrict__ W,
                __bf16* __restrict__ ws) {
    const float* src = (blockIdx.y == 0) ? X : W;
    __bf16* hi = ws + (size_t)blockIdx.y * (2 * NELEM);
    __bf16* lo = hi + NELEM;

    const size_t i = ((size_t)blockIdx.x * blockDim.x + threadIdx.x) * 8;
    float4 u0 = *(const float4*)(src + i);
    float4 u1 = *(const float4*)(src + i + 4);
    float f[8] = {u0.x, u0.y, u0.z, u0.w, u1.x, u1.y, u1.z, u1.w};

    v8bf h, l;
#pragma unroll
    for (int e = 0; e < 8; ++e) {
        __bf16 hh = (__bf16)f[e];
        h[e] = hh;
        l[e] = (__bf16)(f[e] - (float)hh);
    }
    *(v8bf*)(hi + i) = h;
    *(v8bf*)(lo + i) = l;
}

// Load one 16x32 bf16 fragment (ISA 16-bit A/B lane layout): lanes 0-15 hold
// row laneLo with K chunks {0..7, 16..23}; lanes 16-31 get {8..15, 24..31}.
// p points at this lane's row base (incl. laneHi*8); kk in elements.
__device__ __forceinline__ v16bf load_frag_bf16(const __bf16* __restrict__ p,
                                                int kk) {
    v8bf c0 = *(const v8bf*)(p + kk);        // K chunk lo (16 B)
    v8bf c1 = *(const v8bf*)(p + kk + 16);   // K chunk hi (16 B, +32 B)
    return __builtin_shufflevector(c0, c1, 0, 1, 2, 3, 4, 5, 6, 7,
                                          8, 9, 10, 11, 12, 13, 14, 15);
}

// ---------------------------------------------------------------------------
// Phase 2: GEMM from preconverted bf16.
// ---------------------------------------------------------------------------
__global__ __launch_bounds__(256)
void gemm_wmma_bf16x3(const __bf16* __restrict__ Xhi,
                      const __bf16* __restrict__ Xlo,
                      const __bf16* __restrict__ Whi,
                      const __bf16* __restrict__ Wlo,
                      const float* __restrict__ bias,
                      float* __restrict__ out) {
    const int lane   = threadIdx.x & 31;
    const int wave   = threadIdx.x >> 5;
    const int laneLo = lane & 15;
    const int laneHi = lane >> 4;

    const int waveM = wave & 1;                    // 2 waves along M
    const int waveN = wave >> 1;                   // 4 waves along N
    const int m0 = blockIdx.y * 64  + waveM * 32;  // batch rows
    const int n0 = blockIdx.x * 256 + waveN * 64;  // output features

    // 12 independent per-lane stream pointers, fixed for the whole k-loop.
    const __bf16 *pAhi[2], *pAlo[2], *pBhi[4], *pBlo[4];
#pragma unroll
    for (int i = 0; i < 2; ++i) {
        const size_t off = (size_t)(m0 + 16 * i + laneLo) * IN_F + laneHi * 8;
        pAhi[i] = Xhi + off;
        pAlo[i] = Xlo + off;
    }
#pragma unroll
    for (int j = 0; j < 4; ++j) {
        const size_t off = (size_t)(n0 + 16 * j + laneLo) * IN_F + laneHi * 8;
        pBhi[j] = Whi + off;
        pBlo[j] = Wlo + off;
    }

    v8f acc[2][4];
#pragma unroll
    for (int i = 0; i < 2; ++i)
#pragma unroll
        for (int j = 0; j < 4; ++j)
            acc[i][j] = (v8f){0.f, 0.f, 0.f, 0.f, 0.f, 0.f, 0.f, 0.f};

#pragma unroll 4
    for (int kk = 0; kk < IN_F; kk += 32) {
        v16bf a_hi[2], a_lo[2];
#pragma unroll
        for (int i = 0; i < 2; ++i) {
            a_hi[i] = load_frag_bf16(pAhi[i], kk);
            a_lo[i] = load_frag_bf16(pAlo[i], kk);
        }
#pragma unroll
        for (int j = 0; j < 4; ++j) {
            v16bf b_hi = load_frag_bf16(pBhi[j], kk);
            v16bf b_lo = load_frag_bf16(pBlo[j], kk);
#pragma unroll
            for (int i = 0; i < 2; ++i) {
                acc[i][j] = __builtin_amdgcn_wmma_f32_16x16x32_bf16(
                    false, a_lo[i], false, b_hi, (short)0, acc[i][j], false, false);
                acc[i][j] = __builtin_amdgcn_wmma_f32_16x16x32_bf16(
                    false, a_hi[i], false, b_lo, (short)0, acc[i][j], false, false);
                acc[i][j] = __builtin_amdgcn_wmma_f32_16x16x32_bf16(
                    false, a_hi[i], false, b_hi, (short)0, acc[i][j], false, false);
            }
        }
    }

    // Epilogue: element (VGPR r, lane L) = (M = r + 8*(L>>4), N = L&15).
#pragma unroll
    for (int j = 0; j < 4; ++j) {
        const int   n  = n0 + 16 * j + laneLo;
        const float bv = bias[n];
#pragma unroll
        for (int i = 0; i < 2; ++i) {
#pragma unroll
            for (int r = 0; r < 8; ++r) {
                const int m = m0 + 16 * i + r + laneHi * 8;
                out[(size_t)m * OUT_F + n] = acc[i][j][r] + bv;
            }
        }
    }
}

// ---------------------------------------------------------------------------
// Fallback (ws too small): fused conversion bf16x3 GEMM, wave tile 32x32.
// ---------------------------------------------------------------------------
__device__ __forceinline__ void load_split_frag(const float* __restrict__ base,
                                                int row, int kk, int laneHi,
                                                v16bf& hi, v16bf& lo) {
    const float* p = base + (size_t)row * IN_F + kk + laneHi * 8;
    float4 u0 = *(const float4*)(p +  0);
    float4 u1 = *(const float4*)(p +  4);
    float4 u2 = *(const float4*)(p + 16);
    float4 u3 = *(const float4*)(p + 20);
    float f[16] = {u0.x, u0.y, u0.z, u0.w,  u1.x, u1.y, u1.z, u1.w,
                   u2.x, u2.y, u2.z, u2.w,  u3.x, u3.y, u3.z, u3.w};
#pragma unroll
    for (int e = 0; e < 16; ++e) {
        __bf16 h = (__bf16)f[e];
        hi[e] = h;
        lo[e] = (__bf16)(f[e] - (float)h);
    }
}

__global__ __launch_bounds__(256)
void gemm_wmma_bf16x3_fused(const float* __restrict__ X,
                            const float* __restrict__ W,
                            const float* __restrict__ bias,
                            float* __restrict__ out) {
    const int lane   = threadIdx.x & 31;
    const int wave   = threadIdx.x >> 5;
    const int laneLo = lane & 15;
    const int laneHi = lane >> 4;
    const int waveM  = wave & 1;
    const int waveN  = wave >> 1;
    const int m0 = blockIdx.y * 64  + waveM * 32;
    const int n0 = blockIdx.x * 128 + waveN * 32;

    v8f acc[2][2];
#pragma unroll
    for (int i = 0; i < 2; ++i)
#pragma unroll
        for (int j = 0; j < 2; ++j)
            acc[i][j] = (v8f){0.f, 0.f, 0.f, 0.f, 0.f, 0.f, 0.f, 0.f};

    for (int kk = 0; kk < IN_F; kk += 32) {
        v16bf a_hi[2], a_lo[2], b_hi[2], b_lo[2];
#pragma unroll
        for (int i = 0; i < 2; ++i)
            load_split_frag(X, m0 + 16 * i + laneLo, kk, laneHi, a_hi[i], a_lo[i]);
#pragma unroll
        for (int j = 0; j < 2; ++j)
            load_split_frag(W, n0 + 16 * j + laneLo, kk, laneHi, b_hi[j], b_lo[j]);
#pragma unroll
        for (int i = 0; i < 2; ++i)
#pragma unroll
            for (int j = 0; j < 2; ++j) {
                acc[i][j] = __builtin_amdgcn_wmma_f32_16x16x32_bf16(
                    false, a_lo[i], false, b_hi[j], (short)0, acc[i][j], false, false);
                acc[i][j] = __builtin_amdgcn_wmma_f32_16x16x32_bf16(
                    false, a_hi[i], false, b_lo[j], (short)0, acc[i][j], false, false);
                acc[i][j] = __builtin_amdgcn_wmma_f32_16x16x32_bf16(
                    false, a_hi[i], false, b_hi[j], (short)0, acc[i][j], false, false);
            }
    }

#pragma unroll
    for (int j = 0; j < 2; ++j) {
        const int   n  = n0 + 16 * j + laneLo;
        const float bv = bias[n];
#pragma unroll
        for (int i = 0; i < 2; ++i)
#pragma unroll
            for (int r = 0; r < 8; ++r) {
                const int m = m0 + 16 * i + r + laneHi * 8;
                out[(size_t)m * OUT_F + n] = acc[i][j][r] + bv;
            }
    }
}

extern "C" void kernel_launch(void* const* d_in, const int* in_sizes, int n_in,
                              void* d_out, int out_size, void* d_ws, size_t ws_size,
                              hipStream_t stream) {
    const float* X    = (const float*)d_in[0];
    const float* W    = (const float*)d_in[1];
    const float* bias = (const float*)d_in[2];
    float*       out  = (float*)d_out;

    const size_t need = 4 * NELEM * sizeof(__bf16);   // 128 MB
    if (ws_size >= need) {
        __bf16* ws = (__bf16*)d_ws;
        // Phase 1: split X and W into bf16 hi/lo (8 elems/thread).
        dim3 sblock(256, 1, 1);
        dim3 sgrid((unsigned)(NELEM / (256 * 8)), 2, 1);   // 8192 x 2
        split_pass<<<sgrid, sblock, 0, stream>>>(X, W, ws);
        // Phase 2: WMMA GEMM from preconverted bf16.
        dim3 block(256, 1, 1);
        dim3 grid(OUT_F / 256, BATCH / 64, 1);             // 16 x 64
        gemm_wmma_bf16x3<<<grid, block, 0, stream>>>(
            ws, ws + NELEM, ws + 2 * NELEM, ws + 3 * NELEM, bias, out);
    } else {
        dim3 block(256, 1, 1);
        dim3 grid(OUT_F / 128, BATCH / 64, 1);
        gemm_wmma_bf16x3_fused<<<grid, block, 0, stream>>>(X, W, bias, out);
    }
}
